// SNNNetwork_86870008528981
// MI455X (gfx1250) — matrix operand order repaired and checked by reading, not compiled, not run on hardware
//
#include <hip/hip_runtime.h>
#include <hip/hip_bf16.h>

typedef __attribute__((ext_vector_type(16))) _Float16 v16h;
typedef __attribute__((ext_vector_type(8)))  _Float16 v8h;
typedef __attribute__((ext_vector_type(8)))  float    v8f;

#define T_STEPS 500
#define BATCH   32
#define L0N     512
#define L1N     1024
#define L2N     512

// ---------------------------------------------------------------------------
// GLIF neuron update (constants folded: DT*K_M*R*I0 == 1)
// v' = v + x + (a0+a1)/I0 - v*(1/(R*I0)) - s*DT*v
// a'[j] = a[j]*(1-DT*k[j]) + s*DT*(a[j] + amp[j])
// ---------------------------------------------------------------------------
__device__ __forceinline__ float glif_step(float x, float& v, float& a0, float& a1) {
    const float DT      = 0.05f;
    const float INV_RI0 = 9.43f / 700.0f;   // 1/(R*I0)
    const float INV_I0  = 1.0f / 700.0f;
    float s   = 20.0f / (1.0f + __expf(-(v - 10.0f) * 0.1f));
    float sdt = s * DT;
    float vn  = v + x + (a0 + a1) * INV_I0 - v * INV_RI0 - sdt * v;
    float a0n = a0 * 0.99985f + sdt * (a0 - 9.18f);     // k=0.003
    float a1n = a1 * 0.995f   + sdt * (a1 - 198.94f);   // k=0.1
    v = vn; a0 = a0n; a1 = a1n;
    return s;
}

// Scan over time; emit f16 spikes shifted by one step (s_out[t] = s[t-1], s_out[0]=0)
// so the following GEMM consumes the delayed buffer directly.
__global__ void scan_shifted(const float* __restrict__ drive,
                             _Float16* __restrict__ s_out, int nelem) {
    int idx = blockIdx.x * blockDim.x + threadIdx.x;
    if (idx >= nelem) return;
    float v = 0.0f, a0 = 0.0f, a1 = 0.0f;
    s_out[idx] = (_Float16)0.0f;
    for (int t = 0; t < T_STEPS; ++t) {
        float s = glif_step(drive[(size_t)t * nelem + idx], v, a0, a1);
        if (t + 1 < T_STEPS) s_out[(size_t)(t + 1) * nelem + idx] = (_Float16)s;
    }
}

// Final layer: emit f32 spikes unshifted straight into d_out.
__global__ void scan_final(const float* __restrict__ drive,
                           float* __restrict__ out, int nelem) {
    int idx = blockIdx.x * blockDim.x + threadIdx.x;
    if (idx >= nelem) return;
    float v = 0.0f, a0 = 0.0f, a1 = 0.0f;
    for (int t = 0; t < T_STEPS; ++t) {
        out[(size_t)t * nelem + idx] = glif_step(drive[(size_t)t * nelem + idx], v, a0, a1);
    }
}

__global__ void f32_to_f16(const float* __restrict__ src, _Float16* __restrict__ dst, int n) {
    int i = blockIdx.x * blockDim.x + threadIdx.x;
    if (i < n) dst[i] = (_Float16)src[i];
}

// ---------------------------------------------------------------------------
// C[M,N] = A[M,K] * W[N,K]^T, f16 inputs, f32 accumulate via WMMA.
// One wave per 16(M) x 64(N) tile: 1 A-fragment reused across 4 B-fragments.
// M % 16 == 0, N % 64 == 0, K % 32 == 0 (true for all calls here).
// ---------------------------------------------------------------------------
__global__ void __launch_bounds__(256) wmma_gemm_bt(
    const _Float16* __restrict__ A,   // [M,K] row-major
    const _Float16* __restrict__ W,   // [N,K] row-major
    float* __restrict__ C,            // [M,N] row-major
    int M, int N, int K)
{
    const int lane = threadIdx.x & 31;
    const int wave = threadIdx.x >> 5;
    const int gw   = blockIdx.x * 8 + wave;
    const int tilesN = N >> 6;
    const int tilesM = M >> 4;
    if (gw >= tilesM * tilesN) return;
    const int m0 = (gw / tilesN) << 4;
    const int n0 = (gw % tilesN) << 6;

    const int lh = lane & 15;   // row (A) / col (B) within tile
    const int hi = lane >> 4;   // half-wave select

    const _Float16* arow = A + (size_t)(m0 + lh) * K;
    const _Float16* b0p  = W + (size_t)(n0 +  0 + lh) * K + hi * 16;
    const _Float16* b1p  = W + (size_t)(n0 + 16 + lh) * K + hi * 16;
    const _Float16* b2p  = W + (size_t)(n0 + 32 + lh) * K + hi * 16;
    const _Float16* b3p  = W + (size_t)(n0 + 48 + lh) * K + hi * 16;

    v8f acc0 = {}, acc1 = {}, acc2 = {}, acc3 = {};

    for (int kb = 0; kb < K; kb += 32) {
        // A 16x32 fragment: lane holds K {kb+16h..kb+16h+15}
        union { v16h v; v8h h[2]; } a;
        a.h[0] = *reinterpret_cast<const v8h*>(arow + kb + hi * 16);
        a.h[1] = *reinterpret_cast<const v8h*>(arow + kb + hi * 16 + 8);
        // B 32x16 fragments: lane holds 16 contiguous K halves of column n
        v16h b0 = *reinterpret_cast<const v16h*>(b0p + kb);
        v16h b1 = *reinterpret_cast<const v16h*>(b1p + kb);
        v16h b2 = *reinterpret_cast<const v16h*>(b2p + kb);
        v16h b3 = *reinterpret_cast<const v16h*>(b3p + kb);

        acc0 = __builtin_amdgcn_wmma_f32_16x16x32_f16(false, a.v, false, b0, (short)0, acc0, false, false);
        acc1 = __builtin_amdgcn_wmma_f32_16x16x32_f16(false, a.v, false, b1, (short)0, acc1, false, false);
        acc2 = __builtin_amdgcn_wmma_f32_16x16x32_f16(false, a.v, false, b2, (short)0, acc2, false, false);
        acc3 = __builtin_amdgcn_wmma_f32_16x16x32_f16(false, a.v, false, b3, (short)0, acc3, false, false);
    }

    // C/D layout: VGPR i -> row m0 + i + 8*hi, col n0 + 16*nb + lh
    float* crow = C + (size_t)(m0 + hi * 8) * N + n0 + lh;
    #pragma unroll
    for (int i = 0; i < 8; ++i) {
        float* cr = crow + (size_t)i * N;
        cr[0]  = acc0[i];
        cr[16] = acc1[i];
        cr[32] = acc2[i];
        cr[48] = acc3[i];
    }
}

// ---------------------------------------------------------------------------
extern "C" void kernel_launch(void* const* d_in, const int* in_sizes, int n_in,
                              void* d_out, int out_size, void* d_ws, size_t ws_size,
                              hipStream_t stream) {
    const float* inputs = (const float*)d_in[0];   // [500,32,512]
    const float* W1     = (const float*)d_in[1];   // [1024,512]
    const float* W2     = (const float*)d_in[2];   // [512,1024]
    float*       out    = (float*)d_out;           // [500,32,512]

    char* ws = (char*)d_ws;
    const size_t MROWS = (size_t)T_STEPS * BATCH;          // 16000
    _Float16* W1h    = (_Float16*)(ws);                                   // 1 MB
    _Float16* W2h    = (_Float16*)(ws + (1u << 20));                      // 1 MB
    _Float16* s0s    = (_Float16*)(ws + (2u << 20));                      // [T,B,512] f16 (shifted)
    _Float16* s1s    = (_Float16*)((char*)s0s + MROWS * L0N * 2);         // [T,B,1024] f16 (shifted)
    float*    drive1 = (float*)((char*)s1s + MROWS * L1N * 2);            // [T*B,1024] f32
    float*    drive2 = (float*)((char*)drive1 + MROWS * L1N * 4);         // [T*B,512]  f32

    // Weight conversion to f16
    f32_to_f16<<<(L1N * L0N) / 256, 256, 0, stream>>>(W1, W1h, L1N * L0N);
    f32_to_f16<<<(L2N * L1N) / 256, 256, 0, stream>>>(W2, W2h, L2N * L1N);

    // Layer 0: raw input drives the neurons; produce delayed spike history
    scan_shifted<<<(BATCH * L0N) / 256, 256, 0, stream>>>(inputs, s0s, BATCH * L0N);

    // drive1[t*B+b, :] = s0_shift[t*B+b, :] @ W1^T   (M=16000, N=1024, K=512)
    {
        int waves = ((int)MROWS / 16) * (L1N / 64);
        wmma_gemm_bt<<<(waves + 7) / 8, 256, 0, stream>>>(s0s, W1h, drive1,
                                                          (int)MROWS, L1N, L0N);
    }

    // Layer 1 scan -> delayed spike history
    scan_shifted<<<(BATCH * L1N) / 256, 256, 0, stream>>>(drive1, s1s, BATCH * L1N);

    // drive2 = s1_shift @ W2^T   (M=16000, N=512, K=1024)
    {
        int waves = ((int)MROWS / 16) * (L2N / 64);
        wmma_gemm_bt<<<(waves + 7) / 8, 256, 0, stream>>>(s1s, W2h, drive2,
                                                          (int)MROWS, L2N, L1N);
    }

    // Layer 2 scan -> output spikes
    scan_final<<<(BATCH * L2N) / 256, 256, 0, stream>>>(drive2, out, BATCH * L2N);
}
